// GraphConvolution_62620623175771
// MI455X (gfx1250) — matrix operand order repaired and checked by reading, not compiled, not run on hardware
//
#include <hip/hip_runtime.h>
#include <hip/hip_bf16.h>

typedef _Float16 v4h  __attribute__((ext_vector_type(4)));
typedef _Float16 v8h  __attribute__((ext_vector_type(8)));
typedef _Float16 v16h __attribute__((ext_vector_type(16)));
typedef float    v8f  __attribute__((ext_vector_type(8)));
typedef unsigned int u32x4 __attribute__((ext_vector_type(4)));
typedef int          i32x8 __attribute__((ext_vector_type(8)));
typedef int          i32x4 __attribute__((ext_vector_type(4)));

#define D  512         // D_IN == D_OUT == 512
#define TM 128         // block tile M
#define TN 128         // block tile N
#define TK 32          // K step (one wmma K)
#define AS 40          // LDS row stride in halves (32 + 8 pad; matches TDM pad config)

#if __has_builtin(__builtin_amdgcn_tensor_load_to_lds)
#define USE_TDM 1
#endif

// ---------------- fp32 -> fp16 conversion (vectorized) ----------------
__global__ __launch_bounds__(256)
void cvt_f32_f16_kernel(const float* __restrict__ src, _Float16* __restrict__ dst, long n) {
    long i = ((long)blockIdx.x * blockDim.x + threadIdx.x) * 4;
    if (i + 3 < n) {
        float4 v = *(const float4*)(src + i);
        v4h h;
        h[0] = (_Float16)v.x; h[1] = (_Float16)v.y;
        h[2] = (_Float16)v.z; h[3] = (_Float16)v.w;
        *(v4h*)(dst + i) = h;
    }
}

// ---------------- W[k][n] f32 -> Wt[n][k] f16 (transpose once; W is 1 MB) ----------------
__global__ __launch_bounds__(256)
void cvt_w_transpose_kernel(const float* __restrict__ w, _Float16* __restrict__ wt) {
    int i = blockIdx.x * blockDim.x + threadIdx.x;   // 0 .. 512*512-1
    int k = i >> 9;
    int n = i & (D - 1);
    wt[(size_t)n * D + k] = (_Float16)w[i];
}

// ---------------- LDS byte-offset of a __shared__ pointer ----------------
__device__ __forceinline__ unsigned lds_off(const void* p) {
    return (unsigned)(unsigned long long)(__attribute__((address_space(3))) const void*)p;
}

#ifdef USE_TDM
// One TDM 2-D tile load: 32(K) x 128(rows) f16, row stride 512 elems,
// LDS pad 4 DWORDs every 16 DWORDs -> LDS row stride = 40 halves (== AS).
__device__ __forceinline__ void tdm_load_tile(const _Float16* gsrc, unsigned ldsAddr,
                                              unsigned t_d0, unsigned t_d1) {
    unsigned long long ga = (unsigned long long)gsrc;
    u32x4 g0;
    g0.x = 1u;                                        // count=1
    g0.y = ldsAddr;                                   // lds_addr
    g0.z = (unsigned)ga;                              // global_addr lo
    g0.w = (unsigned)((ga >> 32) & 0x01FFFFFFu)       // global_addr hi (57-bit)
         | (2u << 30);                                // type=2 ("image")
    i32x8 g1;
    g1[0] = (int)((1u << 16)                          // data_size = 2 bytes
                | (1u << 20)                          // pad_enable
                | (3u << 22)                          // pad_interval code 3 = 16 DWORDs
                | (3u << 25));                        // pad_amount  code 3 = 4 DWORDs
    g1[1] = (int)((t_d0 & 0xFFFFu) << 16);            // tensor_dim0[15:0]
    g1[2] = (int)(((t_d0 >> 16) & 0xFFFFu)
                | ((t_d1 & 0xFFFFu) << 16));          // tensor_dim1[15:0]
    g1[3] = (int)(((t_d1 >> 16) & 0xFFFFu)
                | ((unsigned)TK << 16));              // tile_dim0 = 32
    g1[4] = TM;                                       // tile_dim1 = 128
    g1[5] = D;                                        // tensor_dim0_stride = 512
    g1[6] = 0;
    g1[7] = 0;
    i32x4 gz = {0, 0, 0, 0};
#if __clang_major__ >= 23
    i32x8 gz8 = {0, 0, 0, 0, 0, 0, 0, 0};
    __builtin_amdgcn_tensor_load_to_lds(g0, g1, gz, gz, gz8, 0);
#else
    __builtin_amdgcn_tensor_load_to_lds(g0, g1, gz, gz, 0);
#endif
}
#endif

// ---------------- WMMA GEMM: suph[N,512](f16) = Xh @ Wt^T ----------------
__global__ __launch_bounds__(256)
void gemm_wmma_kernel(const _Float16* __restrict__ Xh, const _Float16* __restrict__ Wt,
                      _Float16* __restrict__ suph, int NN) {
    __shared__ alignas(16) _Float16 Ah[2][TM * AS];
    __shared__ alignas(16) _Float16 Bh[2][TN * AS];

    const int tid  = threadIdx.x;
    const int lane = tid & 31;
    const int wave = tid >> 5;
    const int wm   = wave & 3;
    const int wn   = wave >> 2;
    const int mBase = blockIdx.y * TM;
    const int nBase = blockIdx.x * TN;

    v8f acc[2][4];
#pragma unroll
    for (int mt = 0; mt < 2; ++mt)
#pragma unroll
        for (int nt = 0; nt < 4; ++nt)
            acc[mt][nt] = (v8f)(0.0f);

    const int arow = lane & 15;
    const int akh  = (lane >> 4) * 8;
    const int bk   = (lane >> 4) * 8;
    const int nSteps = D / TK;

#ifdef USE_TDM
    if (wave == 0) {
        tdm_load_tile(Xh + (size_t)mBase * D, lds_off(&Ah[0][0]), D, (unsigned)(NN - mBase));
        tdm_load_tile(Wt + (size_t)nBase * D, lds_off(&Bh[0][0]), D, (unsigned)(D - nBase));
    }
#endif

    for (int ks = 0; ks < nSteps; ++ks) {
        const int buf = ks & 1;
        const int k0  = ks * TK;
        __syncthreads();                 // all waves done reading buf^1 (iter ks-1)

#ifdef USE_TDM
        if (wave == 0) {
            if (ks + 1 < nSteps) {
                const int k1 = k0 + TK;
                tdm_load_tile(Xh + (size_t)mBase * D + k1, lds_off(&Ah[buf ^ 1][0]),
                              (unsigned)(D - k1), (unsigned)(NN - mBase));
                tdm_load_tile(Wt + (size_t)nBase * D + k1, lds_off(&Bh[buf ^ 1][0]),
                              (unsigned)(D - k1), (unsigned)(D - nBase));
                __builtin_amdgcn_s_wait_tensorcnt(2);   // current buf's 2 DMAs done
            } else {
                __builtin_amdgcn_s_wait_tensorcnt(0);
            }
        }
#else
        for (int c = tid; c < (TM * TK) / 8; c += 256) {
            int row = c >> 2;
            int kh  = (c & 3) * 8;
            int grow = mBase + row; if (grow >= NN) grow = NN - 1;
            *(v8h*)(&Ah[buf][row * AS + kh]) = *(const v8h*)(Xh + (size_t)grow * D + k0 + kh);
            *(v8h*)(&Bh[buf][row * AS + kh]) = *(const v8h*)(Wt + (size_t)(nBase + row) * D + k0 + kh);
        }
#endif
        __syncthreads();                 // tile in buf ready

        v16h a[2], b[4];
#pragma unroll
        for (int mt = 0; mt < 2; ++mt) {
            int r = wm * 32 + mt * 16 + arow;
            v8h lo = *(const v8h*)(&Ah[buf][r * AS + akh]);
            v8h hi = *(const v8h*)(&Ah[buf][r * AS + akh + 16]);
#pragma unroll
            for (int j = 0; j < 8; ++j) { a[mt][j] = lo[j]; a[mt][8 + j] = hi[j]; }
        }
#pragma unroll
        for (int nt = 0; nt < 4; ++nt) {
            int n = wn * 64 + nt * 16 + (lane & 15);
            v8h lo = *(const v8h*)(&Bh[buf][n * AS + bk]);
            v8h hi = *(const v8h*)(&Bh[buf][n * AS + bk + 16]);
#pragma unroll
            for (int j = 0; j < 8; ++j) { b[nt][j] = lo[j]; b[nt][8 + j] = hi[j]; }
        }

#pragma unroll
        for (int mt = 0; mt < 2; ++mt)
#pragma unroll
            for (int nt = 0; nt < 4; ++nt)
                acc[mt][nt] = __builtin_amdgcn_wmma_f32_16x16x32_f16(
                    false, a[mt], false, b[nt], (short)0, acc[mt][nt], false, false);
    }

    // store C as f16 (support stays L2-resident for the gather phase)
    const int cn  = nBase + wn * 64 + (lane & 15);
    const int rhi = (lane >> 4) * 8;
#pragma unroll
    for (int mt = 0; mt < 2; ++mt)
#pragma unroll
        for (int nt = 0; nt < 4; ++nt)
#pragma unroll
            for (int r = 0; r < 8; ++r) {
                int grow = mBase + wm * 32 + mt * 16 + r + rhi;
                if (grow < NN)
                    suph[(size_t)grow * D + cn + nt * 16] = (_Float16)acc[mt][nt][r];
            }
}

// ---------------- CSR build: zero, count, scan, fill (compacting col/val) ----------------
__global__ __launch_bounds__(256)
void zero_kernel(int* __restrict__ p, int n) {
    int i = blockIdx.x * blockDim.x + threadIdx.x;
    if (i < n) p[i] = 0;
}

__global__ __launch_bounds__(256)
void count_kernel(const int* __restrict__ rows, int* __restrict__ counts, int E) {
    int e = blockIdx.x * blockDim.x + threadIdx.x;
    if (e < E) atomicAdd(&counts[rows[e]], 1);       // native global_atomic_add_u32
}

#define SCAN_T 1024
__global__ __launch_bounds__(SCAN_T)
void scan_kernel(const int* __restrict__ counts, int* __restrict__ offsets,
                 int* __restrict__ cursor, int NN) {
    __shared__ int sums[SCAN_T];
    int t = threadIdx.x;
    int chunk = (NN + SCAN_T - 1) / SCAN_T;
    int lo = t * chunk;
    int hi = lo + chunk; if (hi > NN) hi = NN; if (lo > NN) lo = NN;
    int s = 0;
    for (int i = lo; i < hi; ++i) s += counts[i];
    sums[t] = s;
    __syncthreads();
    for (int off = 1; off < SCAN_T; off <<= 1) {     // Hillis-Steele inclusive scan
        int v = (t >= off) ? sums[t - off] : 0;
        __syncthreads();
        sums[t] += v;
        __syncthreads();
    }
    int pre = (t == 0) ? 0 : sums[t - 1];
    for (int i = lo; i < hi; ++i) {
        offsets[i] = pre;
        cursor[i]  = pre;
        pre += counts[i];
    }
    if (t == SCAN_T - 1) offsets[NN] = pre;
}

// Compacts (col, val) into CSR order: gather phase then streams metadata sequentially.
__global__ __launch_bounds__(256)
void fill_kernel(const int* __restrict__ rows, const int* __restrict__ cols,
                 const float* __restrict__ vals, int* __restrict__ cursor,
                 int* __restrict__ colsC, float* __restrict__ valsC, int E) {
    int e = blockIdx.x * blockDim.x + threadIdx.x;
    if (e < E) {
        int pos = atomicAdd(&cursor[rows[e]], 1);
        colsC[pos] = cols[e];
        valsC[pos] = vals[e];
    }
}

// ---------------- CSR gather: out[r] = b + sum val * suph[col]  (no fp atomics) ----------------
// 64 threads x 8 cols: one 16B global_load_b128 of f16 support per lane per edge.
__global__ __launch_bounds__(64)
void gather_kernel(const _Float16* __restrict__ suph, const int* __restrict__ offsets,
                   const int* __restrict__ colsC, const float* __restrict__ valsC,
                   const float* __restrict__ bias, float* __restrict__ out, int NN) {
    int row = blockIdx.x;
    int t = threadIdx.x;                              // 64 threads x 8 cols = 512
    float4 b0 = ((const float4*)bias)[t * 2];
    float4 b1 = ((const float4*)bias)[t * 2 + 1];
    float acc[8] = { b0.x, b0.y, b0.z, b0.w, b1.x, b1.y, b1.z, b1.w };
    int beg = offsets[row], end = offsets[row + 1];
    for (int p = beg; p < end; ++p) {
        int   c = colsC[p];                           // sequential metadata stream
        float v = valsC[p];
        v8h g = ((const v8h*)(suph + (size_t)c * D))[t];   // 16B per lane, L2-resident
#pragma unroll
        for (int j = 0; j < 8; ++j)
            acc[j] += v * (float)g[j];
    }
    float4 o0 = { acc[0], acc[1], acc[2], acc[3] };
    float4 o1 = { acc[4], acc[5], acc[6], acc[7] };
    float4* orow = (float4*)(out + (size_t)row * D);
    orow[t * 2]     = o0;
    orow[t * 2 + 1] = o1;
}

extern "C" void kernel_launch(void* const* d_in, const int* in_sizes, int n_in,
                              void* d_out, int out_size, void* d_ws, size_t ws_size,
                              hipStream_t stream) {
    const float* x       = (const float*)d_in[0];
    const int*   adj_row = (const int*)d_in[1];
    const int*   adj_col = (const int*)d_in[2];
    const float* adj_val = (const float*)d_in[3];
    const float* weight  = (const float*)d_in[4];
    const float* bias    = (const float*)d_in[5];
    float*       out     = (float*)d_out;

    const int  NN = in_sizes[0] / D;              // 100000
    const int  E  = in_sizes[1];                  // 800000
    const long nX = (long)NN * D;
    const long nW = (long)D * D;

    // workspace: [suph f16][Xh f16][Wt f16][counts][offsets NN+1][cursor][colsC E][valsC E]
    _Float16* suph = (_Float16*)d_ws;
    _Float16* Xh   = suph + nX;
    _Float16* Wt   = Xh + nX;
    int*      counts  = (int*)(Wt + nW);
    int*      offsets = counts + NN;
    int*      cursor  = offsets + NN + 1;
    int*      colsC   = cursor + NN;
    float*    valsC   = (float*)(colsC + E);

    // --- CSR build (independent of GEMM) ---
    zero_kernel <<<(NN + 255) / 256, 256, 0, stream>>>(counts, NN);
    count_kernel<<<(E + 255) / 256, 256, 0, stream>>>(adj_row, counts, E);
    scan_kernel <<<1, SCAN_T, 0, stream>>>(counts, offsets, cursor, NN);
    fill_kernel <<<(E + 255) / 256, 256, 0, stream>>>(adj_row, adj_col, adj_val,
                                                      cursor, colsC, valsC, E);

    // --- dense path: convert + WMMA GEMM (TDM double-buffered) ---
    cvt_f32_f16_kernel<<<(int)((nX / 4 + 255) / 256), 256, 0, stream>>>(x, Xh, nX);
    cvt_w_transpose_kernel<<<(int)(nW / 256), 256, 0, stream>>>(weight, Wt);
    dim3 ggrid(D / TN, (NN + TM - 1) / TM);       // (4, 782)
    gemm_wmma_kernel<<<ggrid, 256, 0, stream>>>(Xh, Wt, suph, NN);

    // --- CSR gather with fused bias: one 64-thread block per output row ---
    gather_kernel<<<NN, 64, 0, stream>>>(suph, offsets, colsC, valsC, bias, out, NN);
}
